// LSTMAutoEncoder_13469017440528
// MI455X (gfx1250) — compile-verified, matrix operand order checked by reading, not run on hardware
//
#include <hip/hip_runtime.h>
#include <hip/hip_bf16.h>
#include <math.h>

// Problem constants (B, T, H, F from the reference).
#define BSZ 64
#define TT  512
#define HH  512
#define FF  32
#define GG  2048   // 4*H gate stack

typedef __attribute__((ext_vector_type(16))) _Float16 v16h;
typedef __attribute__((ext_vector_type(8)))  float    v8f;

union AFrag { v16h v; unsigned u[8]; };

// K index of the first element (p=0) of packed pair g for a lane-half,
// per the CDNA5 16-bit A-matrix 16x32 layout (CDNA5 ISA):
//   VGPR0..3: K = 2g+p (+8 for lanes 16-31); VGPR4..7: K = 16+2(g-4)+p (+8).
__device__ __forceinline__ int kk0_of(int g, int hh) {
  return ((g < 4) ? (2 * g) : (16 + 2 * (g - 4))) + 8 * hh;
}

// ---------------------------------------------------------------- utilities
__global__ void zero_words_kernel(unsigned* __restrict__ p, int n) {
  int i = blockIdx.x * blockDim.x + threadIdx.x;
  if (i < n) p[i] = 0u;
}

__global__ void cvt_f16_kernel(const float* __restrict__ s, _Float16* __restrict__ d, int n) {
  int i = blockIdx.x * blockDim.x + threadIdx.x;
  if (i < n) d[i] = (_Float16)s[i];
}

__global__ void bias_sum_kernel(const float* __restrict__ a, const float* __restrict__ b,
                                float* __restrict__ o, int n) {
  int i = blockIdx.x * blockDim.x + threadIdx.x;
  if (i < n) o[i] = a[i] + b[i];
}

// Repack W[Grows, Kdim] (row-major f32, used as the B operand of h@W^T) into
// f16 tiles laid out so lane L of a wave loads 16 contiguous halves:
//   P[((nt*KT + kt)*32 + lane)*16 + e]  with  B[k][n] = W[n][k],
//   n = nt*16 + (lane&15), k = kt*32 + kk(e, lane>>4)   (mirror of A layout).
__global__ void repack_b_kernel(const float* __restrict__ W, _Float16* __restrict__ P,
                                int Grows, int Kdim) {
  int idx = blockIdx.x * blockDim.x + threadIdx.x;
  int total = Grows * Kdim;
  if (idx >= total) return;
  int e    = idx & 15;
  int lane = (idx >> 4) & 31;
  int tile = idx >> 9;              // 32 lanes * 16 halves per tile
  int KT   = Kdim >> 5;
  int kt   = tile % KT;
  int nt   = tile / KT;
  int n    = nt * 16 + (lane & 15);
  int g    = e >> 1, p = e & 1;
  int k    = kt * 32 + kk0_of(g, lane >> 4) + p;
  P[idx] = (_Float16)W[(size_t)n * Kdim + k];
}

// ------------------------------------------------------- fused LSTM step
// Grid: 128 blocks = (4 b-tiles) x (32 hidden-tiles); block: 128 threads (4 waves).
// Wave w computes the 16x16 tile of gate w (columns w*512 + jt*16 ..), i.e.
// gates = x_t @ Wih^T + h @ Whh^T + bias, then the block does the cell update.
__global__ void lstm_step_kernel(const _Float16* __restrict__ x16,   // [B,T,F] f16
                                 int t_src,
                                 const _Float16* __restrict__ pWih,  // packed [G/16][1][32][16]
                                 const _Float16* __restrict__ pWhh,  // packed [G/16][16][32][16]
                                 const float*    __restrict__ bias,  // [G] (bih+bhh)
                                 const _Float16* __restrict__ h_in,  // [B,H] f16
                                 float*          __restrict__ cbuf,  // [B,H] f32, in-place
                                 _Float16*       __restrict__ h_out) // [B,H] f16
{
  __shared__ float sgate[4][16][16];

  const int tid  = threadIdx.x;
  const int wave = tid >> 5;          // gate index 0..3 (i,f,g,o)
  const int lane = tid & 31;
  const int mt   = blockIdx.x >> 5;   // batch tile 0..3
  const int jt   = blockIdx.x & 31;   // hidden tile 0..31
  const int hh   = lane >> 4;
  const int nl   = lane & 15;

  // Accumulator init: broadcast per-column bias.
  const int   ncol = wave * HH + jt * 16 + nl;
  const float bval = bias[ncol];
  v8f acc;
#pragma unroll
  for (int r = 0; r < 8; ++r) acc[r] = bval;

  const int ntile = wave * 32 + jt;   // tile index along G

  // x_t contribution: single K=32 chunk.
  {
    AFrag a, bfr;
    const _Float16* xrow = x16 + ((size_t)(mt * 16 + nl) * TT + t_src) * FF;
#pragma unroll
    for (int g = 0; g < 8; ++g)
      a.u[g] = *(const unsigned*)(xrow + kk0_of(g, hh));
    bfr.v = *(const v16h*)(pWih + ((size_t)ntile * 32 + lane) * 16);
    acc = __builtin_amdgcn_wmma_f32_16x16x32_f16(false, a.v, false, bfr.v,
                                                 (short)0, acc, false, false);
  }

  // h contribution: K = 512 -> 16 chunks of 32.
  const _Float16* hrow  = h_in + (size_t)(mt * 16 + nl) * HH;
  const _Float16* bbase = pWhh + (size_t)ntile * 16 * 32 * 16;
#pragma unroll
  for (int kt = 0; kt < 16; ++kt) {
    AFrag a;
    const _Float16* hp = hrow + kt * 32;
#pragma unroll
    for (int g = 0; g < 8; ++g)
      a.u[g] = *(const unsigned*)(hp + kk0_of(g, hh));
    v16h bv = *(const v16h*)(bbase + ((size_t)kt * 32 + lane) * 16);
    acc = __builtin_amdgcn_wmma_f32_16x16x32_f16(false, a.v, false, bv,
                                                 (short)0, acc, false, false);
  }

  // C/D layout: VGPR r holds M = r + 8*(lane>=16), N = lane&15.
#pragma unroll
  for (int r = 0; r < 8; ++r) sgate[wave][r + 8 * hh][nl] = acc[r];
  __syncthreads();

  // Cell update: 256 elements (16 rows x 16 cols) over 128 threads.
#pragma unroll
  for (int e = 0; e < 2; ++e) {
    int idx = tid * 2 + e;
    int m = idx >> 4;
    int j = idx & 15;
    float ig = sgate[0][m][j];
    float fg = sgate[1][m][j];
    float gg = sgate[2][m][j];
    float og = sgate[3][m][j];
    size_t off = (size_t)(mt * 16 + m) * HH + jt * 16 + j;
    float cv = cbuf[off];
    float si = 1.0f / (1.0f + __expf(-ig));
    float sf = 1.0f / (1.0f + __expf(-fg));
    float so = 1.0f / (1.0f + __expf(-og));
    float c2 = sf * cv + si * tanhf(gg);
    float h2 = so * tanhf(c2);
    cbuf[off]  = c2;
    h_out[off] = (_Float16)h2;
  }
}

// ------------------------------------------------ bulk output projection
// out[b, T-1-i, :] = HA[i] @ Wo^T + bo, rows = i*64 + b, [T*B,512]x[512,32].
// Grid: 256 blocks x 256 threads (8 waves); one 16-row tile per wave, 2 n-tiles.
__global__ void proj_out_kernel(const _Float16* __restrict__ HA,   // [T][B][H] f16
                                const _Float16* __restrict__ pWo,  // packed [2][16][32][16]
                                const float*    __restrict__ bo,   // [F]
                                float*          __restrict__ out)  // [B,T,F] f32
{
  const int tid  = threadIdx.x;
  const int lane = tid & 31;
  const int wave = tid >> 5;
  const int rt   = blockIdx.x * 8 + wave;   // row tile 0..2047
  const int hh   = lane >> 4;
  const int nl   = lane & 15;

  const _Float16* arow = HA + (size_t)(rt * 16 + nl) * HH;

  v8f acc0, acc1;
  const float b0 = bo[nl], b1 = bo[16 + nl];
#pragma unroll
  for (int r = 0; r < 8; ++r) { acc0[r] = b0; acc1[r] = b1; }

#pragma unroll
  for (int kt = 0; kt < 16; ++kt) {
    AFrag a;
    const _Float16* hp = arow + kt * 32;
#pragma unroll
    for (int g = 0; g < 8; ++g)
      a.u[g] = *(const unsigned*)(hp + kk0_of(g, hh));
    v16h bv0 = *(const v16h*)(pWo + ((size_t)(0 * 16 + kt) * 32 + lane) * 16);
    v16h bv1 = *(const v16h*)(pWo + ((size_t)(1 * 16 + kt) * 32 + lane) * 16);
    acc0 = __builtin_amdgcn_wmma_f32_16x16x32_f16(false, a.v, false, bv0,
                                                  (short)0, acc0, false, false);
    acc1 = __builtin_amdgcn_wmma_f32_16x16x32_f16(false, a.v, false, bv1,
                                                  (short)0, acc1, false, false);
  }

#pragma unroll
  for (int r = 0; r < 8; ++r) {
    int row  = rt * 16 + r + 8 * hh;  // = i*64 + b
    int i    = row >> 6;
    int bb   = row & 63;
    int tout = TT - 1 - i;
    size_t o = ((size_t)bb * TT + tout) * FF;
    out[o + nl]      = acc0[r];
    out[o + 16 + nl] = acc1[r];
  }
}

// ---------------------------------------------------------------- launcher
extern "C" void kernel_launch(void* const* d_in, const int* in_sizes, int n_in,
                              void* d_out, int out_size, void* d_ws, size_t ws_size,
                              hipStream_t stream) {
  (void)in_sizes; (void)n_in; (void)out_size; (void)ws_size;

  const float* x     = (const float*)d_in[0];
  const float* eWih0 = (const float*)d_in[1];
  const float* eWhh0 = (const float*)d_in[2];
  const float* ebih0 = (const float*)d_in[3];
  const float* ebhh0 = (const float*)d_in[4];
  const float* dWih0 = (const float*)d_in[9];
  const float* dWhh0 = (const float*)d_in[10];
  const float* dbih0 = (const float*)d_in[11];
  const float* dbhh0 = (const float*)d_in[12];
  const float* Wo    = (const float*)d_in[17];
  const float* bo    = (const float*)d_in[18];
  float* out = (float*)d_out;

  // Bump allocator over workspace (256-byte aligned).
  char* wsb = (char*)d_ws;
  size_t off = 0;
  auto alloc = [&](size_t bytes) -> char* {
    char* p = wsb + off;
    off += (bytes + 255) & ~(size_t)255;
    return p;
  };

  _Float16* x16   = (_Float16*)alloc((size_t)BSZ * TT * FF * 2);
  _Float16* pWihE = (_Float16*)alloc((size_t)GG * FF * 2);
  _Float16* pWhhE = (_Float16*)alloc((size_t)GG * HH * 2);
  _Float16* pWihD = (_Float16*)alloc((size_t)GG * FF * 2);
  _Float16* pWhhD = (_Float16*)alloc((size_t)GG * HH * 2);
  _Float16* pWo   = (_Float16*)alloc((size_t)FF * HH * 2);
  float*    benc  = (float*)alloc((size_t)GG * 4);
  float*    bdec  = (float*)alloc((size_t)GG * 4);
  _Float16* hbuf0 = (_Float16*)alloc((size_t)BSZ * HH * 2);
  _Float16* hbuf1 = (_Float16*)alloc((size_t)BSZ * HH * 2);
  float*    cbuf  = (float*)alloc((size_t)BSZ * HH * 4);
  _Float16* HA    = (_Float16*)alloc((size_t)TT * BSZ * HH * 2);  // hA before step i

  const size_t slab = (size_t)BSZ * HH;  // 32768 elements per hidden slab

  // Init: zero h0 (f16 -> word count) and c0.
  zero_words_kernel<<<(int)((slab * 2 / 4 + 255) / 256), 256, 0, stream>>>((unsigned*)hbuf0, (int)(slab * 2 / 4));
  zero_words_kernel<<<(int)((slab + 255) / 256), 256, 0, stream>>>((unsigned*)cbuf, (int)slab);

  // Precompute: f16 x, packed weights, combined biases.
  {
    int n = BSZ * TT * FF;
    cvt_f16_kernel<<<(n + 255) / 256, 256, 0, stream>>>(x, x16, n);
  }
  repack_b_kernel<<<(GG * FF + 255) / 256, 256, 0, stream>>>(eWih0, pWihE, GG, FF);
  repack_b_kernel<<<(GG * HH + 255) / 256, 256, 0, stream>>>(eWhh0, pWhhE, GG, HH);
  repack_b_kernel<<<(GG * FF + 255) / 256, 256, 0, stream>>>(dWih0, pWihD, GG, FF);
  repack_b_kernel<<<(GG * HH + 255) / 256, 256, 0, stream>>>(dWhh0, pWhhD, GG, HH);
  repack_b_kernel<<<(FF * HH + 255) / 256, 256, 0, stream>>>(Wo, pWo, FF, HH);
  bias_sum_kernel<<<(GG + 255) / 256, 256, 0, stream>>>(ebih0, ebhh0, benc, GG);
  bias_sum_kernel<<<(GG + 255) / 256, 256, 0, stream>>>(dbih0, dbhh0, bdec, GG);

  // Encoder layer 0: only final (h,c) live on. Final h lands in HA[0].
  for (int t = 0; t < TT; ++t) {
    const _Float16* hin = (t & 1) ? hbuf1 : hbuf0;
    _Float16* hout = (t == TT - 1) ? HA : ((t & 1) ? hbuf0 : hbuf1);
    lstm_step_kernel<<<128, 128, 0, stream>>>(x16, t, pWihE, pWhhE, benc, hin, cbuf, hout);
  }

  // Decoder layer 0 over reversed inputs; c continues in place (c0_dec = c_enc final).
  // Only HA[0..T-1] feed the output, so the last cell update (i = T-1) is dead.
  for (int i = 0; i < TT - 1; ++i) {
    lstm_step_kernel<<<128, 128, 0, stream>>>(x16, TT - 1 - i, pWihD, pWhhD, bdec,
                                              HA + (size_t)i * slab, cbuf,
                                              HA + (size_t)(i + 1) * slab);
  }

  // Bulk projection: out[b, T-1-i, :] = HA[i] @ Wo^T + bo.
  proj_out_kernel<<<256, 256, 0, stream>>>(HA, pWo, bo, out);
}
